// ShiTomasiAngleSparseBADWrapper_20298015441057
// MI455X (gfx1250) — matrix operand order repaired and checked by reading, not compiled, not run on hardware
//
#include <hip/hip_runtime.h>
#include <math.h>

// ---- problem constants (match reference) ----
#define BATCH 16
#define HH 1024
#define WW 1024
#define KK 2048
#define PP 256
#define HW (HH*WW)
#define CAP 4096   // candidate capacity for top-K selection (>= K, room for ties)

typedef __attribute__((ext_vector_type(2)))  float    v2f;
typedef __attribute__((ext_vector_type(8)))  float    v8f;
typedef __attribute__((ext_vector_type(16))) _Float16 v16h;
typedef __attribute__((ext_vector_type(4)))  unsigned u32x4;
typedef __attribute__((ext_vector_type(4)))  int      i32x4;
typedef __attribute__((ext_vector_type(8)))  int      i32x8;

// 15-tap Gaussian, sigma=2.5, radius 7, normalized (precomputed literals)
__constant__ float GW[15] = {
  0.00317420f, 0.00898071f, 0.02165110f, 0.04448069f, 0.07787125f,
  0.11617001f, 0.14768137f, 0.15998130f, 0.14768137f, 0.11617001f,
  0.07787125f, 0.04448069f, 0.02165110f, 0.00898071f, 0.00317420f };

// ---- Tensor Data Mover availability / arity detection ----
#if defined(__has_builtin)
#if __has_builtin(__builtin_amdgcn_tensor_load_to_lds)
#define TDM_AVAILABLE 1
#if __has_include(<hip/amd_detail/amd_gfx1250_TDM.h>)
#define TDM_SIX_ARG 1   // therock-10.0 headers present -> 6-arg builtin
#endif
#endif
#endif

#if defined(TDM_AVAILABLE)
// Issue a 2-D TDM tile load: global (dim0 x dim1 remaining extent, elements)
// -> LDS at byte offset lds_off, tile tile0 x tile1 f32 elements, row stride
// `stride` elements, LDS padding of 1 DWORD every (1<<(pad_code+1)) DWORDs.
// D# layout per CDNA5 ISA (groups 0/1; groups 2/3 unused 2D).
__device__ __forceinline__ void tdm_load_2d(unsigned lds_off, const void* gptr,
                                            int dim0, int dim1, int tile0, int tile1,
                                            int stride, int pad_code) {
  unsigned long long ga = (unsigned long long)gptr;
  u32x4 g0;
  g0[0] = 1u;                                            // count=1, no gather
  g0[1] = lds_off;                                       // lds_addr (bytes)
  g0[2] = (unsigned)ga;                                  // global_addr[31:0]
  g0[3] = (unsigned)((ga >> 32) & 0x1FFFFFFull)          // global_addr[56:32]
        | 0x80000000u;                                   // type=2 ("image")
  i32x8 g1;
  g1[0] = (2 << 16) | (1 << 20) | (pad_code << 22);      // data_size=4B, pad_en
  g1[1] = (int)((unsigned)dim0 << 16);                   // tensor_dim0[15:0]
  g1[2] = (int)(((unsigned)dim0 >> 16) | ((unsigned)dim1 << 16));
  g1[3] = (int)(((unsigned)dim1 >> 16) | ((unsigned)tile0 << 16));
  g1[4] = tile1 & 0xFFFF;                                // tile_dim1 (tile_dim2=0)
  g1[5] = stride;                                        // tensor_dim0_stride lo
  g1[6] = 0;
  g1[7] = 0;
  i32x4 z4 = {0, 0, 0, 0};
#if defined(TDM_SIX_ARG)
  i32x8 z8 = {0, 0, 0, 0, 0, 0, 0, 0};
  __builtin_amdgcn_tensor_load_to_lds(g0, g1, z4, z4, z8, 0);
#else
  __builtin_amdgcn_tensor_load_to_lds(g0, g1, z4, z4, 0);
#endif
}
#endif

// ============================================================================
// Kernel 1: fused Sobel + 5x5 box structure tensor -> score map + angle map
// ============================================================================
__global__ __launch_bounds__(256) void maps_kernel(const float* __restrict__ img,
                                                   float* __restrict__ score,
                                                   float* __restrict__ angle) {
  __shared__ float simg[22][23];
  __shared__ float sIx[20][21];
  __shared__ float sIy[20][21];
  int b  = blockIdx.z;
  int y0 = blockIdx.y * 16, x0 = blockIdx.x * 16;
  int tid = threadIdx.y * 16 + threadIdx.x;
  const float* im = img + (size_t)b * HW;

  for (int i = tid; i < 22*22; i += 256) {
    int r = i / 22, c = i % 22;
    int gy = y0 - 3 + r, gx = x0 - 3 + c;
    simg[r][c] = (gy >= 0 && gy < HH && gx >= 0 && gx < WW) ? im[(size_t)gy*WW + gx] : 0.f;
  }
  // CDNA5 prefetch of the next tile row (lowers to global_prefetch_b8)
  if (y0 + 16 < HH) __builtin_prefetch(im + (size_t)(y0 + 16)*WW + x0 + (tid & 15), 0, 1);
  __syncthreads();

  for (int i = tid; i < 20*20; i += 256) {
    int r = i / 20, c = i % 20;
    int gy = y0 - 2 + r, gx = x0 - 2 + c;
    float ix = 0.f, iy = 0.f;
    if (gy >= 0 && gy < HH && gx >= 0 && gx < WW) {  // box filter zero-pads Ix/Iy
      int rr = r + 1, cc = c + 1;
      float a00 = simg[rr-1][cc-1], a01 = simg[rr-1][cc], a02 = simg[rr-1][cc+1];
      float a10 = simg[rr  ][cc-1],                       a12 = simg[rr  ][cc+1];
      float a20 = simg[rr+1][cc-1], a21 = simg[rr+1][cc], a22 = simg[rr+1][cc+1];
      ix = 0.125f * ((a02 + 2.f*a12 + a22) - (a00 + 2.f*a10 + a20));
      iy = 0.125f * ((a20 + 2.f*a21 + a22) - (a00 + 2.f*a01 + a02));
    }
    sIx[r][c] = ix; sIy[r][c] = iy;
  }
  __syncthreads();

  int ty = threadIdx.y, tx = threadIdx.x;
  float sxx = 0.f, syy = 0.f, sxy = 0.f, sx1 = 0.f, sy1 = 0.f;
  for (int dy = 0; dy < 5; ++dy)
    for (int dx = 0; dx < 5; ++dx) {
      float ix = sIx[ty+dy][tx+dx], iy = sIy[ty+dy][tx+dx];
      sxx += ix*ix; syy += iy*iy; sxy += ix*iy; sx1 += ix; sy1 += iy;
    }
  const float inv25 = 1.f/25.f;
  float a = sxx*inv25, bb = syy*inv25, c = sxy*inv25;
  float half_tr = 0.5f*(a + bb);
  float dd = 0.5f*(a - bb);
  float rad = sqrtf(fmaxf(dd*dd + c*c, 0.f));
  size_t o = (size_t)b*HW + (size_t)(y0+ty)*WW + (x0+tx);
  score[o] = half_tr - rad;
  angle[o] = atan2f(sy1, sx1);    // scale of box mean cancels in atan2
}

// ============================================================================
// Kernel 2: horizontal Gaussian as banded GEMM on the f32 matrix pipe.
// out(16x16) = in_ext(16x32) x S(32x16), S[j,n]=g[j-n]; 8x V_WMMA_F32_16X16X4_F32.
// Tile staged into LDS by the Tensor Data Mover (interior tiles); the D# pad
// fields produce the 33-DWORD row stride. Zero-padded taps live in an LDS
// table so fragment builds are branch-free ds_loads.
// ============================================================================
__global__ __launch_bounds__(256) void hblur_kernel(const float* __restrict__ img,
                                                    float* __restrict__ out) {
  __shared__ struct { float gw[48]; float buf[8][16][33]; } sh;
  int tid  = threadIdx.x;
  int lane = tid & 31;
  int w    = tid >> 5;
  int tile = blockIdx.x * 8 + w;
  int tx = tile & 63, ty = (tile >> 6) & 63, b = tile >> 12;
  int x0 = tx * 16, y0 = ty * 16;
  const float* im = img + (size_t)b * HW;

  if (tid < 48) { int t = tid - 16; sh.gw[tid] = (t >= 0 && t < 15) ? GW[t] : 0.f; }

#if defined(TDM_AVAILABLE)
  if (x0 >= 8) {                                  // interior: TDM; OOB right edge
    unsigned loff = (unsigned)(uintptr_t)&sh.buf[w][0][0];   // zero-fills via dim0
    tdm_load_2d(loff, im + (size_t)y0*WW + (x0 - 7),
                WW - (x0 - 7), HH - y0, 32, 16, WW, /*pad 32 dw*/4);
    __builtin_amdgcn_s_wait_tensorcnt(0);
  } else
#endif
  {
    for (int m = 0; m < 16; ++m) {                // edge fallback (zero-pad cols)
      int col = x0 - 7 + lane;
      sh.buf[w][m][lane] = (col >= 0 && col < WW) ? im[(size_t)(y0+m)*WW + col] : 0.f;
    }
  }
  __syncthreads();

  int half = lane >> 4, mn = lane & 15;
  v8f acc = {};
#if __has_builtin(__builtin_amdgcn_wmma_f32_16x16x4_f32)
  for (int s = 0; s < 8; ++s) {
    int j0 = 4*s + 2*half;                        // 32-bit A 16x4 / B 4x16 layout
    v2f af; af.x = sh.buf[w][mn][j0];     af.y = sh.buf[w][mn][j0+1];
    v2f bf; bf.x = sh.gw[j0 - mn + 16];   bf.y = sh.gw[j0 + 1 - mn + 16];
    acc = __builtin_amdgcn_wmma_f32_16x16x4_f32(false, af, false, bf,
                                                (short)0, acc, false, false);
  }
#else
  for (int r = 0; r < 8; ++r) {
    int m = r + 8*half;
    float s2 = 0.f;
    for (int t = 0; t < 15; ++t) s2 += sh.gw[t + 16] * sh.buf[w][m][mn + t];
    acc[r] = s2;
  }
#endif
  float* dst = out + (size_t)b * HW;
  for (int r = 0; r < 8; ++r)                     // C/D layout: VGPR r -> M=r+8*half
    dst[(size_t)(y0 + r + 8*half)*WW + x0 + mn] = acc[r];
}

// ============================================================================
// Kernel 3: vertical Gaussian via V_WMMA_F32_16X16X32_F16.
// out(16x16) = T(16x32)_f16 x tile(32x16)_f16, T[m,j]=g[j-m]. One WMMA per tile.
// Tile staged by TDM (17-DWORD LDS row stride via pad fields).
// ============================================================================
__global__ __launch_bounds__(256) void vblur_kernel(const float* __restrict__ in,
                                                    float* __restrict__ out) {
  __shared__ struct { float gw[48]; float buf[8][32][17]; } sh;
  int tid  = threadIdx.x;
  int lane = tid & 31;
  int w    = tid >> 5;
  int tile = blockIdx.x * 8 + w;
  int tx = tile & 63, ty = (tile >> 6) & 63, b = tile >> 12;
  int x0 = tx * 16, y0 = ty * 16;
  const float* src = in + (size_t)b * HW;

  if (tid < 48) { int t = tid - 16; sh.gw[tid] = (t >= 0 && t < 15) ? GW[t] : 0.f; }

#if defined(TDM_AVAILABLE)
  if (ty > 0) {                                   // interior: TDM; OOB bottom edge
    unsigned loff = (unsigned)(uintptr_t)&sh.buf[w][0][0];   // zero-fills via dim1
    tdm_load_2d(loff, src + (size_t)(y0 - 7)*WW + x0,
                WW - x0, HH - (y0 - 7), 16, 32, WW, /*pad 16 dw*/3);
    __builtin_amdgcn_s_wait_tensorcnt(0);
  } else
#endif
  {
    for (int i = 0; i < 16; ++i) {                // edge fallback (zero-pad rows)
      int t = i*32 + lane;
      int r = t >> 4, c = t & 15;
      int gy = y0 - 7 + r;
      sh.buf[w][r][c] = (gy >= 0 && gy < HH) ? src[(size_t)gy*WW + x0 + c] : 0.f;
    }
  }
  __syncthreads();

  int half = lane >> 4, mn = lane & 15;
  v16h af, bf;
  for (int e = 0; e < 16; ++e) {
    int v = e >> 1, hl = e & 1;                   // 16-bit A 16x32 lane layout
    int kA = ((v < 4) ? 0 : 16) + (half ? 8 : 0) + 2*(v & 3) + hl;
    af[e] = (_Float16)sh.gw[kA - mn + 16];
    int kB = kA;                                  // 16-bit B 32x16 lane layout
    bf[e] = (_Float16)sh.buf[w][kB][mn];
  }
  v8f acc = {};
  acc = __builtin_amdgcn_wmma_f32_16x16x32_f16(false, af, false, bf,
                                               (short)0, acc, false, false);
  float* dst = out + (size_t)b * HW;
  for (int r = 0; r < 8; ++r)
    dst[(size_t)(y0 + r + 8*half)*WW + x0 + mn] = acc[r];
}

// ============================================================================
// Kernel 4: 7x7 NMS + threshold, LDS-tiled with -inf halo.
// ============================================================================
__global__ __launch_bounds__(256) void nms_kernel(const float* __restrict__ score,
                                                  float* __restrict__ sm) {
  __shared__ float ss[22][23];
  int b = blockIdx.z, y0 = blockIdx.y*16, x0 = blockIdx.x*16;
  int tid = threadIdx.y*16 + threadIdx.x;
  const float* s = score + (size_t)b * HW;
  for (int i = tid; i < 22*22; i += 256) {
    int r = i/22, c = i%22;
    int gy = y0-3+r, gx = x0-3+c;
    ss[r][c] = (gy >= 0 && gy < HH && gx >= 0 && gx < WW) ? s[(size_t)gy*WW + gx]
                                                          : -__builtin_inff();
  }
  __syncthreads();
  int ty = threadIdx.y, tx = threadIdx.x;
  float m = -__builtin_inff();
  for (int dy = 0; dy < 7; ++dy)
    for (int dx = 0; dx < 7; ++dx)
      m = fmaxf(m, ss[ty+dy][tx+dx]);
  float v = ss[ty+3][tx+3];
  float r = (v >= m - 1e-7f) ? v : 0.f;
  r = (r > 0.f) ? r : 0.f;
  sm[(size_t)b*HW + (size_t)(y0+ty)*WW + (x0+tx)] = r;
}

// ============================================================================
// Kernel 5: exact per-batch top-K (sorted) via 4-pass MSB radix-select on
// float bit patterns, then LDS bitonic sort of <=4096 survivors.
// ============================================================================
__global__ __launch_bounds__(1024) void select_kernel(const float* __restrict__ sm,
                                                      float* __restrict__ out_kp,
                                                      float* __restrict__ out_sc) {
  __shared__ unsigned hist[256];
  __shared__ unsigned sh_prefix, sh_remaining, sh_count;
  __shared__ float cs[CAP];
  __shared__ int   ci[CAP];
  int b = blockIdx.x;
  int tid = threadIdx.x;
  const float* s = sm + (size_t)b * HW;

  unsigned prefix = 0, remaining = KK;
  for (int pass = 0; pass < 4; ++pass) {
    int shift = 24 - 8*pass;
    if (tid < 256) hist[tid] = 0u;
    __syncthreads();
    unsigned mask = (pass == 0) ? 0u : (0xFFFFFFFFu << (shift + 8));
    for (int i = tid; i < HW; i += 1024) {
      float v = s[i];
      if (v > 0.f) {
        unsigned u = __float_as_uint(v);
        if ((u & mask) == prefix) atomicAdd(&hist[(u >> shift) & 0xFFu], 1u);
      }
    }
    __syncthreads();
    if (tid == 0) {
      unsigned cum = 0; int sel = 0;
      for (int bin = 255; bin >= 0; --bin) {
        unsigned h = hist[bin];
        if (cum + h >= remaining) { sel = bin; break; }
        cum += h;
      }
      sh_prefix    = prefix | ((unsigned)sel << shift);
      sh_remaining = (remaining > cum) ? (remaining - cum) : 1u;
    }
    __syncthreads();
    prefix = sh_prefix; remaining = sh_remaining;
  }
  float tv = __uint_as_float(prefix);            // exact K-th largest value

  if (tid == 0) sh_count = 0u;
  __syncthreads();
  for (int i = tid; i < HW; i += 1024) {
    float v = s[i];
    if (v >= tv && v > 0.f) {
      unsigned p = atomicAdd(&sh_count, 1u);
      if (p < CAP) { cs[p] = v; ci[p] = i; }
    }
  }
  __syncthreads();
  int n = (int)sh_count; if (n > CAP) n = CAP;
  for (int i = tid; i < CAP; i += 1024)
    if (i >= n) { cs[i] = -__builtin_inff(); ci[i] = 0x7FFFFFFF; }
  __syncthreads();

  for (int k2 = 2; k2 <= CAP; k2 <<= 1) {        // bitonic: score desc, idx asc on ties
    for (int j = k2 >> 1; j > 0; j >>= 1) {
      for (int i = tid; i < CAP; i += 1024) {
        int ixj = i ^ j;
        if (ixj > i) {
          float s1 = cs[i], s2 = cs[ixj];
          int   i1 = ci[i], i2 = ci[ixj];
          bool before = (s1 > s2) || (s1 == s2 && i1 < i2);
          if (((i & k2) == 0) ? !before : before) {
            cs[i] = s2; cs[ixj] = s1; ci[i] = i2; ci[ixj] = i1;
          }
        }
      }
      __syncthreads();
    }
  }

  for (int i = tid; i < KK; i += 1024) {
    float v = cs[i]; int idx = ci[i];
    bool valid = (i < n) && (v > 0.f);
    size_t o = (size_t)b*KK + i;
    out_kp[o*2 + 0] = valid ? (float)(idx / WW) : -1.f;
    out_kp[o*2 + 1] = valid ? (float)(idx % WW) : -1.f;
    out_sc[o]       = valid ? v : 0.f;
  }
}

// ============================================================================
// Kernel 6: rotation-aware BAD descriptor. One wave per keypoint, 8 pairs/lane,
// wave32 __shfl_xor reduction for the L2 norm. Gathers stay in the 192 MB L2.
// ============================================================================
__global__ __launch_bounds__(256) void desc_kernel(const float* __restrict__ blur,
                                                   const float* __restrict__ angle,
                                                   const float* __restrict__ pairs,
                                                   const float* __restrict__ out_kp,
                                                   const float* __restrict__ out_sc,
                                                   float* __restrict__ desc) {
  int gwave = (blockIdx.x * 256 + threadIdx.x) >> 5;
  int lane  = threadIdx.x & 31;
  int b = gwave / KK, k = gwave % KK;
  size_t kp = (size_t)b*KK + k;
  float ky = out_kp[kp*2], kx = out_kp[kp*2 + 1];
  float valid = (out_sc[kp] > 0.f) ? 1.f : 0.f;
  int iy = (int)fminf(fmaxf(ky, 0.f), (float)(HH-1));
  int ix = (int)fminf(fmaxf(kx, 0.f), (float)(WW-1));
  const float* bl = blur + (size_t)b * HW;
  float ang = angle[(size_t)b*HW + (size_t)iy*WW + ix];
  float c = cosf(ang), s = sinf(ang);

  float d[8]; float ss = 0.f;
  for (int q = 0; q < 8; ++q) {
    int p = q*32 + lane;
    float dy0 = pairs[p*4+0], dx0 = pairs[p*4+1];
    float dy1 = pairs[p*4+2], dx1 = pairs[p*4+3];
    float ry0 = s*dx0 + c*dy0, rx0 = c*dx0 - s*dy0;
    float ry1 = s*dx1 + c*dy1, rx1 = c*dx1 - s*dy1;
    int sy0 = (int)fminf(fmaxf(rintf(ky + ry0), 0.f), (float)(HH-1));
    int sx0 = (int)fminf(fmaxf(rintf(kx + rx0), 0.f), (float)(WW-1));
    int sy1 = (int)fminf(fmaxf(rintf(ky + ry1), 0.f), (float)(HH-1));
    int sx1 = (int)fminf(fmaxf(rintf(kx + rx1), 0.f), (float)(WW-1));
    float v0 = bl[(size_t)sy0*WW + sx0];
    float v1 = bl[(size_t)sy1*WW + sx1];
    float dd = 1.f / (1.f + __expf(-10.f*(v0 - v1)));
    d[q] = dd; ss += dd*dd;
  }
  for (int off = 16; off > 0; off >>= 1) ss += __shfl_xor(ss, off, 32);
  float inv = valid / fmaxf(sqrtf(ss), 1e-8f);
  float* dp = desc + kp*PP;
  for (int q = 0; q < 8; ++q) dp[q*32 + lane] = d[q] * inv;
}

// ============================================================================
// Host launcher. Workspace layout (floats): score | angle | tmp(hblur->sm) | blur
// ============================================================================
extern "C" void kernel_launch(void* const* d_in, const int* in_sizes, int n_in,
                              void* d_out, int out_size, void* d_ws, size_t ws_size,
                              hipStream_t stream) {
  (void)in_sizes; (void)n_in; (void)out_size; (void)ws_size;
  const float* img   = (const float*)d_in[0];   // (B,1,H,W) f32
  const float* pairs = (const float*)d_in[1];   // (P,2,2)   f32

  float* out      = (float*)d_out;
  float* out_kp   = out;                                  // (B,K,2)
  float* out_sc   = out + (size_t)BATCH*KK*2;             // (B,K)
  float* out_desc = out_sc + (size_t)BATCH*KK;            // (B,K,P)

  const size_t N = (size_t)BATCH * HW;
  float* ws    = (float*)d_ws;
  float* score = ws;
  float* angle = ws + N;
  float* tmp   = ws + 2*N;   // hblur output; reused as NMS output (stream-ordered)
  float* blur  = ws + 3*N;

  dim3 tgrid(WW/16, HH/16, BATCH), tblk(16, 16);
  int tiles = BATCH * (HH/16) * (WW/16);

  maps_kernel  <<<tgrid, tblk, 0, stream>>>(img, score, angle);
  hblur_kernel <<<tiles/8, 256, 0, stream>>>(img, tmp);
  vblur_kernel <<<tiles/8, 256, 0, stream>>>(tmp, blur);
  nms_kernel   <<<tgrid, tblk, 0, stream>>>(score, tmp);
  select_kernel<<<BATCH, 1024, 0, stream>>>(tmp, out_kp, out_sc);
  desc_kernel  <<<(BATCH*KK)/8, 256, 0, stream>>>(blur, angle, pairs,
                                                  out_kp, out_sc, out_desc);
}
